// FusedSparseLMHead_85856396247757
// MI455X (gfx1250) — matrix operand order repaired and checked by reading, not compiled, not run on hardware
//
#include <hip/hip_runtime.h>
#include <math.h>

// Problem constants (from reference): B*S = 4096 tokens, D = 2048, V = 65536
#define N_TOK   4096
#define DIM     2048
#define VOCAB   65536
#define SLABS   16
#define SLAB_V  (VOCAB / SLABS)     // 4096 vocab rows per slab
#define TILE_M  128                 // tokens per workgroup
#define TILE_NV 256                 // vocab cols per inner tile
#define KBLK    32                  // K block staged in LDS (one fp16 WMMA K-step)
#define SBF     40                  // padded LDS row stride (halfs) for A/B stages
#define LDL     129                 // padded LDS row stride (floats) for logits tile
#define NT      (N_TOK / TILE_M)    // 32 token tiles

// LDS byte offsets (dynamic LDS starts at offset 0 of the wave's LDS_BASE).
// Double-buffered fp16 stages + logits half-tile.
#define LDS_A0  0u
#define LDS_B0  10240u              // A buffer size = 128*40*2
#define LDS_A1  30720u              // B buffer size = 256*40*2
#define LDS_B1  40960u
#define LDS_L   61440u
#define SMEM_BYTES (61440u + (unsigned)(TILE_M * LDL * 4))   // 127488 B -> 2 blocks/WGP

typedef __attribute__((ext_vector_type(4)))  _Float16 v4h;
typedef __attribute__((ext_vector_type(8)))  _Float16 v8h;
typedef __attribute__((ext_vector_type(16))) _Float16 v16h;
typedef __attribute__((ext_vector_type(8)))  float    v8f;

// ---------------------------------------------------------------------------
// One-time fp32 -> fp16 conversion (streamed once; halves GEMM-side traffic)
// ---------------------------------------------------------------------------
__global__ __launch_bounds__(256)
void cvt_f32_f16_kernel(const float* __restrict__ src, _Float16* __restrict__ dst, int n4)
{
    for (int i = blockIdx.x * blockDim.x + threadIdx.x; i < n4;
         i += gridDim.x * blockDim.x) {
        const float4 v = ((const float4*)src)[i];
        v4h p = { (_Float16)v.x, (_Float16)v.y, (_Float16)v.z, (_Float16)v.w };
        ((v4h*)dst)[i] = p;
    }
}

// ---------------------------------------------------------------------------
// Async memory->LDS staging of one K-block (GLOBAL_LOAD_ASYNC_TO_LDS_B128,
// tracked by ASYNCcnt; completes in order -> enables partial waits).
// ---------------------------------------------------------------------------
__device__ __forceinline__
void stage_async(const _Float16* __restrict__ hH, const _Float16* __restrict__ WH,
                 int tokenBase, int vocBase, int kb, int tid,
                 unsigned aOff, unsigned bOff)
{
    // A: 128 rows x 4 chunks of 8 halfs = 512 slots (2/thread)
    #pragma unroll
    for (int it = 0; it < 2; ++it) {
        const int slot = tid + it * 256;
        const int row  = slot >> 2;
        const int c8   = slot & 3;
        const unsigned voff = (unsigned)(((size_t)(tokenBase + row) * DIM + kb + c8 * 8) * 2);
        const unsigned lds  = aOff + (unsigned)((row * SBF + c8 * 8) * 2);
        asm volatile("global_load_async_to_lds_b128 %0, %1, %2"
                     :: "v"(lds), "v"(voff), "s"(hH) : "memory");
    }
    // B: 256 rows x 4 chunks = 1024 slots (4/thread)
    #pragma unroll
    for (int it = 0; it < 4; ++it) {
        const int slot = tid + it * 256;
        const int row  = slot >> 2;
        const int c8   = slot & 3;
        const unsigned voff = (unsigned)(((size_t)(vocBase + row) * DIM + kb + c8 * 8) * 2);
        const unsigned lds  = bOff + (unsigned)((row * SBF + c8 * 8) * 2);
        asm volatile("global_load_async_to_lds_b128 %0, %1, %2"
                     :: "v"(lds), "v"(voff), "s"(WH) : "memory");
    }
}

// ---------------------------------------------------------------------------
// One 16x16x32 fp16 WMMA K-step over the staged buffers.
// A frag layout (ISA 7.12.2): lanes 0-15: K 0-7 then K 16-23;
// lanes 16-31: K 8-15 then K 24-31. B frag: same K order as A, col n = lane%16.
// ---------------------------------------------------------------------------
__device__ __forceinline__
void wmma_kstep(const char* smem, unsigned aOff, unsigned bOff,
                v8f (&acc)[4][4], int wrow, int wcol, int half, int l16)
{
    const _Float16* As = (const _Float16*)(smem + aOff);
    const _Float16* Bs = (const _Float16*)(smem + bOff);
    v16h a[4];
    #pragma unroll
    for (int i = 0; i < 4; ++i) {
        const _Float16* ar = &As[(wrow * 64 + i * 16 + l16) * SBF];
        v8h alo = *(const v8h*)&ar[8 * half];
        v8h ahi = *(const v8h*)&ar[16 + 8 * half];
        a[i] = __builtin_shufflevector(alo, ahi,
                0,1,2,3,4,5,6,7,8,9,10,11,12,13,14,15);
    }
    #pragma unroll
    for (int j = 0; j < 4; ++j) {
        const _Float16* br = &Bs[(wcol * 64 + j * 16 + l16) * SBF];
        v8h blo = *(const v8h*)&br[8 * half];
        v8h bhi = *(const v8h*)&br[16 + 8 * half];
        const v16h b = __builtin_shufflevector(blo, bhi,
                0,1,2,3,4,5,6,7,8,9,10,11,12,13,14,15);
        #pragma unroll
        for (int i = 0; i < 4; ++i)
            acc[i][j] = __builtin_amdgcn_wmma_f32_16x16x32_f16(
                false, a[i], false, b, (short)0, acc[i][j], false, false);
    }
}

// ---------------------------------------------------------------------------
// Phase 1: per (token, vocab-slab) online logsumexp partials.
// FP16IN=true: operands pre-converted in d_ws, double-buffered async staging.
// FP16IN=false: fp32 operands, batched loads + v_cvt_pk_f16_f32 staging.
// ---------------------------------------------------------------------------
template<bool FP16IN>
__global__ __launch_bounds__(256)
void ce_partial_kernel(const float*    __restrict__ hF,
                       const float*    __restrict__ WF,
                       const _Float16* __restrict__ hH,
                       const _Float16* __restrict__ WH,
                       const int*      __restrict__ labels,
                       float* __restrict__ m_part,
                       float* __restrict__ s_part,
                       float* __restrict__ lab_part)
{
    extern __shared__ char smem[];
    float* Ls = (float*)(smem + LDS_L);        // [TILE_M][LDL] logits half-tile

    const int tid  = threadIdx.x;
    const int lane = tid & 31;
    const int wave = tid >> 5;
    const int half = lane >> 4;                // 0: lanes 0-15, 1: lanes 16-31
    const int l16  = lane & 15;
    const int wrow = wave >> 2;                // 0..1 -> row tiles wrow*4 + {0..3}
    const int wcol = wave & 3;                 // 0..3 -> col tiles wcol*4 + {0..3}

    const int tokenBase = blockIdx.x * TILE_M;
    const int slab      = blockIdx.y;
    const int slabBase  = slab * SLAB_V;

    // Online logsumexp state for the row owned by this thread (tid < 128)
    float m_run = -INFINITY;
    float s_run = 0.0f;
    float lab_run = 0.0f;
    int   myLabel = -1;
    if (tid < TILE_M) myLabel = labels[tokenBase + tid];

    for (int vt = 0; vt < SLAB_V / TILE_NV; ++vt) {
        const int vocBase = slabBase + vt * TILE_NV;

        v8f acc[4][4];
        #pragma unroll
        for (int i = 0; i < 4; ++i)
            #pragma unroll
            for (int j = 0; j < 4; ++j) {
                v8f z = {0.f,0.f,0.f,0.f,0.f,0.f,0.f,0.f};
                acc[i][j] = z;
            }

        if constexpr (FP16IN) {
            // Double-buffered async pipeline. DIM/KBLK = 64 (even), so the
            // 2x-unrolled loop is exact and kb2+KBLK is always in range.
            stage_async(hH, WH, tokenBase, vocBase, 0, tid, LDS_A0, LDS_B0);
            for (int kb2 = 0; kb2 < DIM; kb2 += 2 * KBLK) {
                stage_async(hH, WH, tokenBase, vocBase, kb2 + KBLK, tid, LDS_A1, LDS_B1);
                // ASYNCcnt completes in order: <=6 outstanding => buf0's 6 done
                asm volatile("s_wait_asynccnt 6" ::: "memory");
                __syncthreads();
                wmma_kstep(smem, LDS_A0, LDS_B0, acc, wrow, wcol, half, l16);
                __syncthreads();
                if (kb2 + 2 * KBLK < DIM) {
                    stage_async(hH, WH, tokenBase, vocBase, kb2 + 2 * KBLK, tid, LDS_A0, LDS_B0);
                    asm volatile("s_wait_asynccnt 6" ::: "memory");
                } else {
                    asm volatile("s_wait_asynccnt 0" ::: "memory");
                }
                __syncthreads();
                wmma_kstep(smem, LDS_A1, LDS_B1, acc, wrow, wcol, half, l16);
                __syncthreads();
            }
        } else {
            _Float16* As = (_Float16*)(smem + LDS_A0);
            _Float16* Bs = (_Float16*)(smem + LDS_B0);
            for (int kb = 0; kb < DIM; kb += KBLK) {
                __syncthreads();   // previous fragment reads done before restage
                // Batch ALL loads first (keeps 12 loads in flight), then cvt+store.
                float4 ta[4];
                #pragma unroll
                for (int it = 0; it < 4; ++it) {
                    const int slot = tid + it * 256;
                    const int row  = slot >> 3;
                    const int c4   = slot & 7;
                    ta[it] = *(const float4*)&hF[(size_t)(tokenBase + row) * DIM + kb + c4 * 4];
                }
                float4 tb[8];
                #pragma unroll
                for (int it = 0; it < 8; ++it) {
                    const int slot = tid + it * 256;
                    const int row  = slot >> 3;
                    const int c4   = slot & 7;
                    tb[it] = *(const float4*)&WF[(size_t)(vocBase + row) * DIM + kb + c4 * 4];
                }
                #pragma unroll
                for (int it = 0; it < 4; ++it) {
                    const int slot = tid + it * 256;
                    const int row  = slot >> 3;
                    const int c4   = slot & 7;
                    v4h p = { (_Float16)ta[it].x, (_Float16)ta[it].y,
                              (_Float16)ta[it].z, (_Float16)ta[it].w };
                    *(v4h*)&As[row * SBF + c4 * 4] = p;
                }
                #pragma unroll
                for (int it = 0; it < 8; ++it) {
                    const int slot = tid + it * 256;
                    const int row  = slot >> 3;
                    const int c4   = slot & 7;
                    v4h p = { (_Float16)tb[it].x, (_Float16)tb[it].y,
                              (_Float16)tb[it].z, (_Float16)tb[it].w };
                    *(v4h*)&Bs[row * SBF + c4 * 4] = p;
                    if (kb + KBLK < DIM) {
                        __builtin_prefetch(&WF[(size_t)(vocBase + row) * DIM + kb + KBLK + c4 * 4], 0, 0);
                    }
                }
                __syncthreads();
                wmma_kstep(smem, LDS_A0, LDS_B0, acc, wrow, wcol, half, l16);
            }
            __syncthreads();   // fragment reads done before Ls spill below
        }

        // Two 128-column passes: spill acc -> LDS logits, then online update.
        // C/D layout: VGPR v: lanes 0-15 -> M=v, N=lane; lanes 16-31 -> M=v+8.
        #pragma unroll
        for (int pass = 0; pass < 2; ++pass) {
            __syncthreads();
            if ((wcol >> 1) == pass) {
                #pragma unroll
                for (int i = 0; i < 4; ++i) {
                    const int rtg = wrow * 4 + i;
                    #pragma unroll
                    for (int j = 0; j < 4; ++j) {
                        const int ctl = (wcol & 1) * 4 + j;   // 0..7 within pass
                        #pragma unroll
                        for (int v = 0; v < 8; ++v)
                            Ls[(rtg * 16 + v + 8 * half) * LDL + ctl * 16 + l16] = acc[i][j][v];
                    }
                }
            }
            __syncthreads();
            if (tid < TILE_M) {
                const int colBase = vocBase + pass * 128;
                const float* lrow = &Ls[tid * LDL];
                float cmax = -INFINITY;
                #pragma unroll 8
                for (int c = 0; c < 128; ++c) cmax = fmaxf(cmax, lrow[c]);
                const float m_new = fmaxf(m_run, cmax);
                float s = s_run * __expf(m_run - m_new);
                #pragma unroll 8
                for (int c = 0; c < 128; ++c) s += __expf(lrow[c] - m_new);
                m_run = m_new;
                s_run = s;
                const int idx = myLabel - colBase;
                if (idx >= 0 && idx < 128) lab_run += lrow[idx];
            }
        }
        __syncthreads();
    }

    if (tid < TILE_M) {
        const int token = tokenBase + tid;
        m_part[slab * N_TOK + token]   = m_run;
        s_part[slab * N_TOK + token]   = s_run;
        lab_part[slab * N_TOK + token] = lab_run;
    }
}

// ---------------------------------------------------------------------------
// Phase 2: merge slab partials -> per-token NLL -> weighted mean (scalar)
// ---------------------------------------------------------------------------
__global__ __launch_bounds__(256)
void ce_finalize_kernel(const float* __restrict__ m_part,
                        const float* __restrict__ s_part,
                        const float* __restrict__ lab_part,
                        const float* __restrict__ weights,
                        float* __restrict__ out)
{
    __shared__ float red_num[256];
    __shared__ float red_den[256];
    const int tid = threadIdx.x;
    float num = 0.0f, den = 0.0f;
    for (int t = tid; t < N_TOK; t += 256) {
        float m = -INFINITY;
        #pragma unroll
        for (int c = 0; c < SLABS; ++c) m = fmaxf(m, m_part[c * N_TOK + t]);
        float s = 0.0f, lab = 0.0f;
        #pragma unroll
        for (int c = 0; c < SLABS; ++c) {
            s   += s_part[c * N_TOK + t] * __expf(m_part[c * N_TOK + t] - m);
            lab += lab_part[c * N_TOK + t];
        }
        const float nll = m + logf(s) - lab;
        const float w = weights[t];
        num += w * nll;
        den += w;
    }
    red_num[tid] = num;
    red_den[tid] = den;
    __syncthreads();
    for (int off = 128; off > 0; off >>= 1) {
        if (tid < off) {
            red_num[tid] += red_num[tid + off];
            red_den[tid] += red_den[tid + off];
        }
        __syncthreads();
    }
    if (tid == 0) out[0] = red_num[0] / red_den[0];
}

// ---------------------------------------------------------------------------
extern "C" void kernel_launch(void* const* d_in, const int* in_sizes, int n_in,
                              void* d_out, int out_size, void* d_ws, size_t ws_size,
                              hipStream_t stream) {
    const float* h      = (const float*)d_in[0];   // hidden_states [4096, 2048]
    const int*   labels = (const int*)  d_in[1];   // labels        [4096]
    const float* w      = (const float*)d_in[2];   // loss_weights  [4096]
    const float* W      = (const float*)d_in[3];   // W             [65536, 2048]

    const size_t PART_BYTES = (size_t)3 * SLABS * N_TOK * sizeof(float);   // 768 KB
    const size_t H16_BYTES  = (size_t)N_TOK * DIM * sizeof(_Float16);      // 16 MB
    const size_t W16_BYTES  = (size_t)VOCAB * DIM * sizeof(_Float16);      // 256 MB

    float* ws       = (float*)d_ws;
    float* m_part   = ws;
    float* s_part   = ws + (size_t)SLABS * N_TOK;
    float* lab_part = ws + (size_t)2 * SLABS * N_TOK;

    dim3 grid(NT, SLABS);  // x fastest => 32 consecutive blocks share one W slab (L2 reuse)

    const bool pre = ws_size >= (PART_BYTES + H16_BYTES + W16_BYTES);
    if (pre) {
        _Float16* h16 = (_Float16*)((char*)d_ws + PART_BYTES);
        _Float16* W16 = (_Float16*)((char*)d_ws + PART_BYTES + H16_BYTES);
        hipLaunchKernelGGL(cvt_f32_f16_kernel, dim3(1024), dim3(256), 0, stream,
                           h, h16, (int)(N_TOK * DIM / 4));
        hipLaunchKernelGGL(cvt_f32_f16_kernel, dim3(8192), dim3(256), 0, stream,
                           W, W16, (int)(VOCAB * (size_t)DIM / 4));
        hipLaunchKernelGGL((ce_partial_kernel<true>), grid, dim3(256), SMEM_BYTES, stream,
                           (const float*)nullptr, (const float*)nullptr, h16, W16,
                           labels, m_part, s_part, lab_part);
    } else {
        hipLaunchKernelGGL((ce_partial_kernel<false>), grid, dim3(256), SMEM_BYTES, stream,
                           h, W, (const _Float16*)nullptr, (const _Float16*)nullptr,
                           labels, m_part, s_part, lab_part);
    }
    hipLaunchKernelGGL(ce_finalize_kernel, dim3(1), dim3(256), 0, stream,
                       m_part, s_part, lab_part, w, (float*)d_out);
}
